// VectorQuantizer_5669356830824
// MI455X (gfx1250) — matrix operand order repaired and checked by reading, not compiled, not run on hardware
//
#include <hip/hip_runtime.h>
#include <hip/hip_bf16.h>
#include <math.h>

typedef __attribute__((ext_vector_type(16))) _Float16 v16h;
typedef __attribute__((ext_vector_type(8)))  _Float16 v8h;
typedef __attribute__((ext_vector_type(8)))  float    v8f;

#define BB    32
#define CC    256
#define HWN   1024          // 32*32
#define NTOK  32768         // BB*HWN
#define KK    1024
#define NELEM 8388608       // BB*CC*HWN

#define CHUNK_CODES 32                  // codes staged per LDS chunk
#define CHUNK_HALF  (CHUNK_CODES * CC)  // 8192 halves = 16 KB
#define NCHUNK      (KK / CHUNK_CODES)  // 32

// ---- gfx1250 async global->LDS (ASYNCcnt-tracked) ----
__device__ __forceinline__ void async_b128_to_lds(unsigned lds_addr,
                                                  unsigned long long gaddr) {
    asm volatile("global_load_async_to_lds_b128 %0, %1, off"
                 :: "v"(lds_addr), "v"(gaddr) : "memory");
}
__device__ __forceinline__ void wait_async0() {
#if __has_builtin(__builtin_amdgcn_s_wait_asynccnt)
    __builtin_amdgcn_s_wait_asynccnt(0);
#else
    asm volatile("s_wait_asynccnt 0" ::: "memory");
#endif
}

// ---------------------------------------------------------------- zero scratch
__global__ void k_zero(float* __restrict__ sums, int* __restrict__ hist) {
    int t = blockIdx.x * blockDim.x + threadIdx.x;
    if (t == 0) sums[0] = 0.0f;
    if (t < KK) hist[t] = 0;
}

// ------------------------------------------- codebook: f32->f16 + row norms
__global__ __launch_bounds__(256)
void k_prep_cb(const float* __restrict__ cb, _Float16* __restrict__ cb_h,
               float* __restrict__ cnorm) {
    int k = blockIdx.x;      // one code row per block
    int c = threadIdx.x;
    float v = cb[(size_t)k * CC + c];
    cb_h[(size_t)k * CC + c] = (_Float16)v;
    float s = v * v;
    #pragma unroll
    for (int off = 16; off; off >>= 1) s += __shfl_xor(s, off, 32);
    __shared__ float ls[8];
    int wave = threadIdx.x >> 5, lane = threadIdx.x & 31;
    if (lane == 0) ls[wave] = s;
    __syncthreads();
    if (threadIdx.x == 0) {
        float t = 0.0f;
        #pragma unroll
        for (int i = 0; i < 8; i++) t += ls[i];
        cnorm[k] = t;
    }
}

// --------------------- z [B,C,HW] f32 -> z_h [B*HW, C] f16 (tiled transpose)
__global__ __launch_bounds__(256)
void k_prep_z(const float* __restrict__ z, _Float16* __restrict__ z_h) {
    __shared__ float tile[32][33];
    int ct = blockIdx.x, ht = blockIdx.y, b = blockIdx.z;
    int tx = threadIdx.x, ty = threadIdx.y;          // block (32,8)
    const float* zb = z + (size_t)b * CC * HWN;
    #pragma unroll
    for (int i = 0; i < 4; i++) {
        int c  = ct * 32 + ty + 8 * i;
        int hw = ht * 32 + tx;
        tile[ty + 8 * i][tx] = zb[(size_t)c * HWN + hw];
    }
    __syncthreads();
    _Float16* ob = z_h + (size_t)b * HWN * CC;
    #pragma unroll
    for (int i = 0; i < 4; i++) {
        int hw = ht * 32 + ty + 8 * i;
        int c  = ct * 32 + tx;
        ob[(size_t)hw * CC + c] = (_Float16)tile[tx][ty + 8 * i];
    }
}

// -------- fused WMMA distance + argmin, codebook double-buffered in LDS
//          via async global->LDS; 8 waves/WG share each staged chunk.
//          A-tile in registers; B software-pipelined one k-step ahead.
__global__ __launch_bounds__(256)
void k_argmin(const _Float16* __restrict__ z_h, const _Float16* __restrict__ cb_h,
              const float* __restrict__ cnorm, int* __restrict__ d_idx) {
    __shared__ _Float16 smem[2][CHUNK_HALF];   // 2 x 16 KB

    int tid  = threadIdx.x;
    int wave = tid >> 5;
    int lane = tid & 31;
    int l    = lane & 15;        // position within 16-lane group
    int hs   = lane >> 4;        // half select
    int row0 = blockIdx.x * 128 + wave * 16;

    const _Float16* arow = z_h + (size_t)(row0 + l) * CC;

    // ---- preload entire A tile (16 tokens x 256 ch) into registers: 64 VGPRs
    v16h areg[8];
    #pragma unroll
    for (int j = 0; j < 8; ++j) {
        ((v8h*)&areg[j])[0] = *(const v8h*)(arow + j * 32 + hs * 16);
        ((v8h*)&areg[j])[1] = *(const v8h*)(arow + j * 32 + hs * 16 + 8);
    }

    float minval[8];
    int   minidx[8];
    #pragma unroll
    for (int r = 0; r < 8; r++) { minval[r] = 3.0e38f; minidx[r] = 0; }

    // --- prologue: stage chunk 0 ---
    {
        unsigned lds0 = (unsigned)(size_t)(&smem[0][0]);
        unsigned long long g0 = (unsigned long long)(size_t)cb_h;
        #pragma unroll
        for (int i = 0; i < 4; i++) {
            unsigned off = (unsigned)(i * 4096 + tid * 16);
            async_b128_to_lds(lds0 + off, g0 + off);
        }
    }
    wait_async0();
    __syncthreads();

    int buf = 0;
    for (int ch = 0; ch < NCHUNK; ++ch) {
        // issue next chunk into the other buffer (overlaps with WMMA below)
        if (ch + 1 < NCHUNK) {
            unsigned ldsn = (unsigned)(size_t)(&smem[buf ^ 1][0]);
            unsigned long long gn =
                (unsigned long long)(size_t)(cb_h + (size_t)(ch + 1) * CHUNK_HALF);
            #pragma unroll
            for (int i = 0; i < 4; i++) {
                unsigned off = (unsigned)(i * 4096 + tid * 16);
                async_b128_to_lds(ldsn + off, gn + off);
            }
        }
        // compute 2 code tiles from the resident chunk, B pipelined 1 step
        const _Float16* brow0 = &smem[buf][0] + (size_t)l * CC;
        const _Float16* brow1 = brow0 + (size_t)16 * CC;
        v8f acc0 = {}, acc1 = {};
        v16h b0 = *(const v16h*)(brow0 + hs * 16);
        v16h b1 = *(const v16h*)(brow1 + hs * 16);
        #pragma unroll
        for (int j = 0; j < 8; ++j) {
            v16h nb0 = {}, nb1 = {};
            if (j + 1 < 8) {
                nb0 = *(const v16h*)(brow0 + (j + 1) * 32 + hs * 16);
                nb1 = *(const v16h*)(brow1 + (j + 1) * 32 + hs * 16);
            }
            acc0 = __builtin_amdgcn_wmma_f32_16x16x32_f16(
                false, areg[j], false, b0, (short)0, acc0, false, false);
            acc1 = __builtin_amdgcn_wmma_f32_16x16x32_f16(
                false, areg[j], false, b1, (short)0, acc1, false, false);
            b0 = nb0; b1 = nb1;
        }
        int n0 = ch * CHUNK_CODES;
        float cn0 = cnorm[n0 + l];
        float cn1 = cnorm[n0 + 16 + l];
        #pragma unroll
        for (int r = 0; r < 8; r++) {
            float d0 = cn0 - 2.0f * acc0[r];   // |e|^2 - 2 z.e
            float d1 = cn1 - 2.0f * acc1[r];
            if (d0 < minval[r]) { minval[r] = d0; minidx[r] = n0 + l; }
            if (d1 < minval[r]) { minval[r] = d1; minidx[r] = n0 + 16 + l; }
        }
        wait_async0();        // next chunk fully in LDS (this wave's share)
        __syncthreads();      // all waves done reading + all shares landed
        buf ^= 1;
    }

    // butterfly min over each 16-lane group; lowest index wins ties
    #pragma unroll
    for (int r = 0; r < 8; r++) {
        float v = minval[r]; int ix = minidx[r];
        #pragma unroll
        for (int off = 8; off >= 1; off >>= 1) {
            float ov = __shfl_xor(v, off, 32);
            int   oi = __shfl_xor(ix, off, 32);
            if (ov < v || (ov == v && oi < ix)) { v = ov; ix = oi; }
        }
        if (l == 0) d_idx[row0 + hs * 8 + r] = ix;   // token (r, hs) of tile
    }
}

// --------------------- gather z_q (BCHW), accumulate sum((z_q - z)^2)
__global__ __launch_bounds__(256)
void k_gather(const float* __restrict__ z, const float* __restrict__ cb,
              const int* __restrict__ d_idx, float* __restrict__ out,
              float* __restrict__ sums) {
    size_t o  = (size_t)blockIdx.x * blockDim.x + threadIdx.x;   // < NELEM
    int b   = (int)(o >> 18);          // / (CC*HWN)
    int rem = (int)(o & 262143);
    int c   = rem >> 10;
    int hw  = rem & 1023;
    int n   = (b << 10) | hw;
    int k   = d_idx[n];
    float q  = cb[(size_t)k * CC + c];
    float zv = z[o];
    out[o] = q;                         // z + sg(z_q - z) == z_q
    float d = q - zv;
    float s = d * d;
    #pragma unroll
    for (int off = 16; off; off >>= 1) s += __shfl_xor(s, off, 32);
    __shared__ float ls[8];
    int wave = threadIdx.x >> 5, lane = threadIdx.x & 31;
    if (lane == 0) ls[wave] = s;
    __syncthreads();
    if (threadIdx.x == 0) {
        float t = 0.0f;
        #pragma unroll
        for (int i = 0; i < 8; i++) t += ls[i];
        atomicAdd(sums, t);
    }
}

// --------------------------------------------------------- usage histogram
__global__ void k_hist(const int* __restrict__ d_idx, int* __restrict__ hist) {
    int n = blockIdx.x * blockDim.x + threadIdx.x;
    atomicAdd(&hist[d_idx[n]], 1);
}

// ------------------------------------------- vq_loss + perplexity scalars
__global__ __launch_bounds__(1024)
void k_final(const float* __restrict__ sums, const int* __restrict__ hist,
             float* __restrict__ out_scalars) {
    int k = threadIdx.x;                 // 1024 threads
    float p = (float)hist[k] / (float)NTOK;
    p = fmaxf(p, 1e-10f);
    float e = p * logf(p);
    #pragma unroll
    for (int off = 16; off; off >>= 1) e += __shfl_xor(e, off, 32);
    __shared__ float ls[32];
    int wave = threadIdx.x >> 5, lane = threadIdx.x & 31;
    if (lane == 0) ls[wave] = e;
    __syncthreads();
    if (threadIdx.x == 0) {
        float t = 0.0f;
        #pragma unroll
        for (int i = 0; i < 32; i++) t += ls[i];
        out_scalars[0] = 1.25f * sums[0] / (float)NELEM;   // (1+beta)*MSE
        out_scalars[1] = expf(-t);                         // perplexity
    }
}

extern "C" void kernel_launch(void* const* d_in, const int* in_sizes, int n_in,
                              void* d_out, int out_size, void* d_ws, size_t ws_size,
                              hipStream_t stream) {
    const float* z  = (const float*)d_in[0];   // [32,256,32,32]
    const float* cb = (const float*)d_in[1];   // [1024,256]
    float* out = (float*)d_out;                // [NELEM + 2]

    char* w = (char*)d_ws;
    _Float16* z_h  = (_Float16*)w;  w += (size_t)NTOK * CC * sizeof(_Float16);
    _Float16* cb_h = (_Float16*)w;  w += (size_t)KK * CC * sizeof(_Float16);
    float*    cnorm = (float*)w;    w += (size_t)KK * sizeof(float);
    int*      idx   = (int*)w;      w += (size_t)NTOK * sizeof(int);
    int*      hist  = (int*)w;      w += (size_t)KK * sizeof(int);
    float*    sums  = (float*)w;    w += 256;

    k_zero   <<<dim3(4),            dim3(256),     0, stream>>>(sums, hist);
    k_prep_cb<<<dim3(KK),           dim3(256),     0, stream>>>(cb, cb_h, cnorm);
    k_prep_z <<<dim3(8, 32, 32),    dim3(32, 8),   0, stream>>>(z, z_h);
    k_argmin <<<dim3(NTOK / 128),   dim3(256),     0, stream>>>(z_h, cb_h, cnorm, idx);
    k_gather <<<dim3(NELEM / 256),  dim3(256),     0, stream>>>(z, cb, idx, out, sums);
    k_hist   <<<dim3(NTOK / 256),   dim3(256),     0, stream>>>(idx, hist);
    k_final  <<<dim3(1),            dim3(1024),    0, stream>>>(sums, hist, out + NELEM);
}
